// CantileverPINN_77318001262885
// MI455X (gfx1250) — matrix operand order repaired and checked by reading, not compiled, not run on hardware
//
#include <hip/hip_runtime.h>
#include <hip/hip_bf16.h>

typedef __attribute__((ext_vector_type(16))) _Float16 v16h;
typedef __attribute__((ext_vector_type(8)))  _Float16 v8h;
typedef __attribute__((ext_vector_type(8)))  float    v8f;

#define WPB      8           // waves per block (wave32)
#define TPB      (WPB * 32)  // 256 threads
#define NBLOCKS  512

// Branch-free tanh: tanh(u) = 1 - 2/(exp(2u)+1), via raw v_exp_f32 / v_rcp_f32.
// Saturates correctly: exp->inf => 1, exp->0 => -1. No exec-mask churn.
__device__ __forceinline__ float tanh_fast(float u)
{
    const float LOG2E_X2 = 2.8853900817779268f;   // 2*log2(e)
    float e2 = __builtin_amdgcn_exp2f(u * LOG2E_X2);
    return 1.0f - 2.0f * __builtin_amdgcn_rcpf(e2 + 1.0f);
}

// tanh derivative chain at t = tanh(u)
__device__ __forceinline__ void tanh_derivs(float t, float& s1, float& s2, float& s3, float& s4)
{
    s1 = 1.0f - t * t;                        // tanh'
    s2 = -2.0f * t * s1;                      // tanh''
    s3 = -2.0f * (s1 * s1 + t * s2);          // tanh'''
    s4 = -2.0f * (3.0f * s1 * s2 + t * s3);   // tanh''''
}

// 4th-order tanh jet (general input jet u0..u4), Faa di Bruno.
__device__ __forceinline__ void tanh_jet(float u0, float u1, float u2, float u3, float u4,
                                         float& y0, float& y1, float& y2, float& y3, float& y4)
{
    float t, s1, s2, s3, s4;
    t = tanh_fast(u0);
    tanh_derivs(t, s1, s2, s3, s4);
    float u1sq = u1 * u1;
    y0 = t;
    y1 = s1 * u1;
    y2 = s2 * u1sq + s1 * u2;
    y3 = s3 * u1sq * u1 + 3.0f * s2 * u1 * u2 + s1 * u3;
    y4 = s4 * u1sq * u1sq + 6.0f * s3 * u1sq * u2
       + s2 * (3.0f * u2 * u2 + 4.0f * u1 * u3) + s1 * u4;
}

// Specialized jet for layer 1 where the input jet is (u, w, 0, 0, 0): y_k = s_k * w^k.
__device__ __forceinline__ void tanh_jet_l1(float u0, float w,
                                            float& y0, float& y1, float& y2, float& y3, float& y4)
{
    float t, s1, s2, s3, s4;
    t = tanh_fast(u0);
    tanh_derivs(t, s1, s2, s3, s4);
    float w2 = w * w;
    y0 = t;
    y1 = s1 * w;
    y2 = s2 * w2;
    y3 = s3 * w2 * w;
    y4 = s4 * w2 * w2;
}

__global__ __launch_bounds__(TPB) void pinn_jet_wmma_kernel(
    const float* __restrict__ x,
    const float* __restrict__ W1, const float* __restrict__ b1,
    const float* __restrict__ W2, const float* __restrict__ b2,
    const float* __restrict__ W3, const float* __restrict__ b3,
    const float* __restrict__ W4, const float* __restrict__ b4,
    float* __restrict__ out, int n, int ntiles)
{
    // per-wave private f16 staging for the D->A tile transpose (layer2 -> layer3)
    __shared__ _Float16 __align__(16) smem[WPB][5][16][32];

    const int lane = threadIdx.x & 31;
    const int lo   = lane & 15;   // column / row-in-tile index
    const int hi   = lane >> 4;   // half-wave select
    const int warp = threadIdx.x >> 5;
    _Float16 (*my)[16][32] = smem[warp];

    // ------- build f16 B fragments for W2 (32x32 padded) and W3 (32x64 padded) -------
    // B fragment layout (16x16x32, f16): lane holds column n = lo; element e<8 holds K = hi*8 + e, e>=8 holds K = 16 + hi*8 + (e-8).
    v16h B2[2], B3[4];
    #pragma unroll
    for (int nt = 0; nt < 2; ++nt) {
        const int nc = nt * 16 + lo;
        #pragma unroll
        for (int e = 0; e < 16; ++e) {
            const int k = (e < 8) ? (hi * 8 + e) : (16 + hi * 8 + (e - 8));
            const float w = (k < 15 && nc < 30) ? W2[k * 30 + nc] : 0.0f;
            B2[nt][e] = (_Float16)w;
        }
    }
    #pragma unroll
    for (int nt = 0; nt < 4; ++nt) {
        const int nc = nt * 16 + lo;
        #pragma unroll
        for (int e = 0; e < 16; ++e) {
            const int k = (e < 8) ? (hi * 8 + e) : (16 + hi * 8 + (e - 8));
            const float w = (k < 30 && nc < 60) ? W3[k * 60 + nc] : 0.0f;
            B3[nt][e] = (_Float16)w;
        }
    }
    // biases keyed by the D-layout column this lane owns in each N-tile
    float bias2[2], bias3[4], w4v[4];
    #pragma unroll
    for (int nt = 0; nt < 2; ++nt) { const int nc = nt * 16 + lo; bias2[nt] = (nc < 30) ? b2[nc] : 0.0f; }
    #pragma unroll
    for (int nt = 0; nt < 4; ++nt) {
        const int nc = nt * 16 + lo;
        bias3[nt] = (nc < 60) ? b3[nc] : 0.0f;
        w4v[nt]   = (nc < 60) ? W4[nc] : 0.0f;  // W4 is (60,1)
    }
    const float b4s = b4[0];

    // layer-1 per-lane weights: half-wave 0 owns units 0..7, half-wave 1 owns 8..15 (15 invalid -> 0)
    const int ub = hi * 8;
    float w1t[8], b1t[8];
    #pragma unroll
    for (int t = 0; t < 8; ++t) {
        const int u = ub + t;
        w1t[t] = (u < 15) ? W1[u] : 0.0f;
        b1t[t] = (u < 15) ? b1[u] : 0.0f;
    }

    const int gwave  = blockIdx.x * WPB + warp;
    const int nwaves = (int)gridDim.x * WPB;

    for (int tile = gwave; tile < ntiles; tile += nwaves) {
        const int pbase = tile * 16;
        const int xi    = pbase + lo;
        const float xv  = x[(xi < n) ? xi : (n - 1)];

        // ---------- layer 1 (VALU) directly into A-fragment layout ----------
        // A layout (16x16x32 f16): row = lane%16; element e<8 -> K = hi*8 + e (our unit index),
        // elements 8..15 -> K in [16,32) = zero padding.
        v16h A[5];
        #pragma unroll
        for (int c = 0; c < 5; ++c)
            #pragma unroll
            for (int e = 0; e < 16; ++e) A[c][e] = (_Float16)0.0f;
        #pragma unroll
        for (int t = 0; t < 8; ++t) {
            const float w  = w1t[t];
            const float u0 = xv * w + b1t[t];   // jet: (u0, w, 0, 0, 0); w==0 for pad units -> all-zero jet
            float y0, y1, y2, y3, y4;
            tanh_jet_l1(u0, w, y0, y1, y2, y3, y4);
            A[0][t] = (_Float16)y0; A[1][t] = (_Float16)y1; A[2][t] = (_Float16)y2;
            A[3][t] = (_Float16)y3; A[4][t] = (_Float16)y4;
        }

        // ---------- layer 2: WMMA (5 channels x 2 N-tiles), tanh jets, stage to LDS ----------
        #pragma unroll
        for (int nt = 0; nt < 2; ++nt) {
            v8f D[5];
            #pragma unroll
            for (int c = 0; c < 5; ++c) {
                v8f cin;
                const float bv = (c == 0) ? bias2[nt] : 0.0f;  // bias only in the value channel
                #pragma unroll
                for (int r = 0; r < 8; ++r) cin[r] = bv;
                D[c] = __builtin_amdgcn_wmma_f32_16x16x32_f16(
                    false, A[c], false, B2[nt], (short)0, cin, false, false);
            }
            const int col = nt * 16 + lo;
            #pragma unroll
            for (int r = 0; r < 8; ++r) {
                float y0, y1, y2, y3, y4;
                tanh_jet(D[0][r], D[1][r], D[2][r], D[3][r], D[4][r], y0, y1, y2, y3, y4);
                const int row = r + 8 * hi;     // D layout: VGPR r, half hi -> matrix row
                my[0][row][col] = (_Float16)y0;
                my[1][row][col] = (_Float16)y1;
                my[2][row][col] = (_Float16)y2;
                my[3][row][col] = (_Float16)y3;
                my[4][row][col] = (_Float16)y4;
            }
        }
        __builtin_amdgcn_wave_barrier();        // keep LDS stores before loads (same wave, in-order DS)
        asm volatile("" ::: "memory");

        // ---------- reload as layer-3 A fragments (tile transpose via LDS) ----------
        v16h A3[5];
        #pragma unroll
        for (int c = 0; c < 5; ++c) {
            const v8h q0 = *(const v8h*)&my[c][lo][hi * 8];        // K = hi*8 .. +7
            const v8h q1 = *(const v8h*)&my[c][lo][16 + hi * 8];   // K = 16+hi*8 .. +7
            #pragma unroll
            for (int e = 0; e < 8; ++e) { A3[c][e] = q0[e]; A3[c][8 + e] = q1[e]; }
        }
        __builtin_amdgcn_wave_barrier();
        asm volatile("" ::: "memory");

        // ---------- layer 3: WMMA (5 x 4 N-tiles) + tanh jets + layer-4 contraction ----------
        float acc[5][8];
        #pragma unroll
        for (int c = 0; c < 5; ++c)
            #pragma unroll
            for (int r = 0; r < 8; ++r) acc[c][r] = 0.0f;

        #pragma unroll
        for (int nt = 0; nt < 4; ++nt) {
            v8f D[5];
            #pragma unroll
            for (int c = 0; c < 5; ++c) {
                v8f cin;
                const float bv = (c == 0) ? bias3[nt] : 0.0f;
                #pragma unroll
                for (int r = 0; r < 8; ++r) cin[r] = bv;
                D[c] = __builtin_amdgcn_wmma_f32_16x16x32_f16(
                    false, A3[c], false, B3[nt], (short)0, cin, false, false);
            }
            const float wj = w4v[nt];   // W4 weight for the hidden column this lane owns
            #pragma unroll
            for (int r = 0; r < 8; ++r) {
                float y0, y1, y2, y3, y4;
                tanh_jet(D[0][r], D[1][r], D[2][r], D[3][r], D[4][r], y0, y1, y2, y3, y4);
                acc[0][r] = fmaf(y0, wj, acc[0][r]);
                acc[1][r] = fmaf(y1, wj, acc[1][r]);
                acc[2][r] = fmaf(y2, wj, acc[2][r]);
                acc[3][r] = fmaf(y3, wj, acc[3][r]);
                acc[4][r] = fmaf(y4, wj, acc[4][r]);
            }
        }

        // ---------- reduce over the 16 hidden-column lanes of each half-wave ----------
        #pragma unroll
        for (int c = 0; c < 5; ++c)
            #pragma unroll
            for (int r = 0; r < 8; ++r) {
                float v = acc[c][r];
                v += __shfl_xor(v, 1, 16);
                v += __shfl_xor(v, 2, 16);
                v += __shfl_xor(v, 4, 16);
                v += __shfl_xor(v, 8, 16);
                acc[c][r] = v;
            }

        // ---------- write out[c*N + point]; one writer lane per (c,r) per half ----------
        #pragma unroll
        for (int c = 0; c < 5; ++c) {
            #pragma unroll
            for (int r = 0; r < 8; ++r) {
                const int p = pbase + r + 8 * hi;
                const float v = acc[c][r] + ((c == 0) ? b4s : 0.0f);
                if (lo == ((c * 8 + r) & 15) && p < n)
                    out[c * n + p] = v;
            }
        }
    }
}

extern "C" void kernel_launch(void* const* d_in, const int* in_sizes, int n_in,
                              void* d_out, int out_size, void* d_ws, size_t ws_size,
                              hipStream_t stream) {
    const float* x   = (const float*)d_in[0];
    const float* W1v = (const float*)d_in[1];
    const float* b1v = (const float*)d_in[2];
    const float* W2v = (const float*)d_in[3];
    const float* b2v = (const float*)d_in[4];
    const float* W3v = (const float*)d_in[5];
    const float* b3v = (const float*)d_in[6];
    const float* W4v = (const float*)d_in[7];
    const float* b4v = (const float*)d_in[8];
    float* out = (float*)d_out;

    const int n      = in_sizes[0];
    const int ntiles = (n + 15) / 16;

    hipLaunchKernelGGL(pinn_jet_wmma_kernel, dim3(NBLOCKS), dim3(TPB), 0, stream,
                       x, W1v, b1v, W2v, b2v, W3v, b3v, W4v, b4v, out, n, ntiles);
}
